// TtShiftedWindowAttention_63677185131154
// MI455X (gfx1250) — matrix operand order repaired and checked by reading, not compiled, not run on hardware
//
#include <hip/hip_runtime.h>
#include <hip/hip_bf16.h>

typedef __attribute__((ext_vector_type(16))) __bf16        v16bf;
typedef __attribute__((ext_vector_type(8)))  float         v8f;
typedef __attribute__((ext_vector_type(4)))  unsigned int  v4u;

#define BB      16
#define HH      56
#define WWID    56
#define CC      256
#define WS      7
#define SHIFT   3
#define HEADS   8
#define NTOK    49
#define NPAD    64
#define HD      32
#define NWIN    64
#define C3      768

// ---------------- WMMA fragment loaders (CDNA5 wave32 layouts) ----------------
// All loaders issue exactly two 16-byte LDS loads (ds_load_b128).

__device__ inline v16bf load_a_frag(const __bf16* base, int ld, int m0, int k0, int lane) {
  int m  = m0 + (lane & 15);
  int kb = (lane >> 4) * 16;
  const char* p = (const char*)base + ((size_t)m * ld + k0 + kb) * 2;
  union { v4u q[2]; v16bf b; } f;
  f.q[0] = *(const v4u*)p;
  f.q[1] = *(const v4u*)(p + 16);
  return f.b;
}

__device__ inline v16bf load_bT_frag(const __bf16* base, int ld, int n0, int c0, int lane) {
  int n  = n0 + (lane & 15);
  int kb = (lane >> 4) * 16;
  const char* p = (const char*)base + ((size_t)n * ld + c0 + kb) * 2;
  union { v4u q[2]; v16bf b; } f;
  f.q[0] = *(const v4u*)p;
  f.q[1] = *(const v4u*)(p + 16);
  return f.b;
}

__device__ inline v16bf load_b_swz(const __bf16* base, int tile, int lane) {
  const char* p = (const char*)base + ((size_t)tile * 32 + lane) * 32;
  union { v4u q[2]; v16bf b; } f;
  f.q[0] = *(const v4u*)p;
  f.q[1] = *(const v4u*)(p + 16);
  return f.b;
}

// Swizzled B-tile element index for element (k, j) of a K x 64 block stored as
// (ktile, ntile) 32x16 fragment tiles.
__device__ inline int b_swz_idx(int k, int j) {
  int kt = k >> 5, krem = k & 31;
  int lh = krem >> 4, e = krem & 15;
  int nt = j >> 4, n = j & 15;
  int tile = kt * 4 + nt;
  return (tile * 32 + (n + lh * 16)) * 16 + e;
}

// ------------------------------- kernel ---------------------------------------
// LDS layout (bytes):
//   [0,      32K)  sQ  64x256 bf16 (row-major)
//   [32K,    64K)  sK  64x256 bf16 (row-major)
//   [64K,    96K)  sV  fragment-swizzled: tile ((h*2+nt)*2+kt) of 32x16
//   [96K,   128K)  sO  64x256 bf16 (row-major)
//   [128K,  192K)  scratch: A/B: sX 64x256 bf16 @128K, sW swizzled 32KB @160K
//                  C   : sS 64x64 f32 @128K, sP 64x64 bf16 @144K
#define SMEM_BYTES 196608

__global__ __launch_bounds__(256) void swin_shifted_window_attn_kernel(
    const float* __restrict__ x, const float* __restrict__ qkv_w,
    const float* __restrict__ qkv_b, const float* __restrict__ proj_w,
    const float* __restrict__ proj_b, const float* __restrict__ rel_bias,
    const float* __restrict__ attn_mask, float* __restrict__ out)
{
  extern __shared__ char smem[];
  __bf16* sQ = (__bf16*)(smem);
  __bf16* sK = (__bf16*)(smem + 32768);
  __bf16* sV = (__bf16*)(smem + 65536);
  __bf16* sO = (__bf16*)(smem + 98304);
  __bf16* sX = (__bf16*)(smem + 131072);
  float*  sS = (float*) (smem + 131072);
  __bf16* sP = (__bf16*)(smem + 147456);
  __bf16* sW = (__bf16*)(smem + 163840);

  const int tid  = threadIdx.x;
  const int lane = tid & 31;
  const int wave = tid >> 5;
  const int blk  = blockIdx.x;
  const int bat  = blk >> 6;
  const int win  = blk & (NWIN - 1);
  const int nh   = win >> 3;
  const int nw   = win & 7;

  // Per-wave GEMM tile assignment: one m-tile, two adjacent n-tiles
  const int mt  = wave >> 1;
  const int nt0 = (wave & 1) * 2;

  // ---------------- Phase A: gather shifted window into LDS as bf16 ----------
  for (int idx = tid; idx < NPAD * CC; idx += 256) {
    int m = idx >> 8, ch = idx & (CC - 1);
    float v = 0.0f;
    if (m < NTOK) {
      int r  = m / WS, c = m - r * WS;
      int hs = nh * WS + r, ws = nw * WS + c;
      int hg = hs + SHIFT; if (hg >= HH)   hg -= HH;
      int wg = ws + SHIFT; if (wg >= WWID) wg -= WWID;
      v = x[(((size_t)bat * HH + hg) * WWID + wg) * CC + ch];
    }
    sX[idx] = (__bf16)v;
  }
  __syncthreads();

  // ---------------- Phase B: QKV = Xw @ qkv_w + b (bf16 WMMA) ----------------
  const float qscale = 0.17677669529663687f;  // 32^-0.5
  for (int nb = 0; nb < C3 / 64; ++nb) {
    // stage 256x64 weight block, pre-swizzled into B-fragment order
    for (int idx = tid; idx < CC * 64; idx += 256) {
      int k = idx >> 6, j = idx & 63;
      sW[b_swz_idx(k, j)] = (__bf16)qkv_w[(size_t)k * C3 + nb * 64 + j];
    }
    __syncthreads();

    v8f acc0 = {}, acc1 = {};
    // software-pipelined k-loop: prefetch k+1 fragments before k's WMMAs
    v16bf a_c  = load_a_frag(sX, CC, mt * 16, 0, lane);
    v16bf b0_c = load_b_swz(sW, nt0, lane);
    v16bf b1_c = load_b_swz(sW, nt0 + 1, lane);
#pragma unroll
    for (int kt = 0; kt < 8; ++kt) {
      v16bf a_n, b0_n, b1_n;
      if (kt < 7) {
        a_n  = load_a_frag(sX, CC, mt * 16, (kt + 1) * 32, lane);
        b0_n = load_b_swz(sW, (kt + 1) * 4 + nt0, lane);
        b1_n = load_b_swz(sW, (kt + 1) * 4 + nt0 + 1, lane);
      }
      acc0 = __builtin_amdgcn_wmma_f32_16x16x32_bf16(false, a_c, false, b0_c,
                                                     (short)0, acc0, false, false);
      acc1 = __builtin_amdgcn_wmma_f32_16x16x32_bf16(false, a_c, false, b1_c,
                                                     (short)0, acc1, false, false);
      a_c = a_n; b0_c = b0_n; b1_c = b1_n;
    }

#pragma unroll
    for (int half = 0; half < 2; ++half) {
      const v8f& acc = half ? acc1 : acc0;
      int nt = nt0 + half;
#pragma unroll
      for (int r = 0; r < 8; ++r) {
        int m  = mt * 16 + r + (lane >> 4) * 8;
        int gn = nb * 64 + nt * 16 + (lane & 15);
        float v = acc[r] + qkv_b[gn];
        if (gn < CC) {
          sQ[m * CC + gn] = (__bf16)(v * qscale);
        } else if (gn < 2 * CC) {
          sK[m * CC + (gn - CC)] = (__bf16)v;
        } else {
          // store V directly in fragment-swizzled order for the P @ V GEMM
          int cv = gn - 2 * CC;
          int h  = cv >> 5, nw2 = cv & 31;
          int ntv = nw2 >> 4, nn = nw2 & 15;
          int ktv = m >> 5,  krem = m & 31;
          int lh  = krem >> 4, e = krem & 15;
          int tile = (h * 2 + ntv) * 2 + ktv;
          sV[(tile * 32 + (nn + lh * 16)) * 16 + e] = (__bf16)v;
        }
      }
    }
    __syncthreads();
  }

  // ---------------- Phase C: per-head attention ------------------------------
  for (int h = 0; h < HEADS; ++h) {
    // S = Q_h @ K_h^T + rel_bias + mask (bias fused into accumulator init)
#pragma unroll
    for (int half = 0; half < 2; ++half) {
      int nt = nt0 + half;
      v8f acc;
#pragma unroll
      for (int r = 0; r < 8; ++r) {
        int m = mt * 16 + r + (lane >> 4) * 8;
        int n = nt * 16 + (lane & 15);
        float init = 0.0f;
        if (m < NTOK && n < NTOK)
          init = rel_bias[((size_t)h * NTOK + m) * NTOK + n] +
                 attn_mask[((size_t)win * NTOK + m) * NTOK + n];
        acc[r] = init;
      }
      v16bf a  = load_a_frag (sQ, CC, mt * 16, h * HD, lane);
      v16bf bm = load_bT_frag(sK, CC, nt * 16, h * HD, lane);
      acc = __builtin_amdgcn_wmma_f32_16x16x32_bf16(false, a, false, bm,
                                                    (short)0, acc, false, false);
#pragma unroll
      for (int r = 0; r < 8; ++r) {
        int m = mt * 16 + r + (lane >> 4) * 8;
        int n = nt * 16 + (lane & 15);
        sS[m * NPAD + n] = acc[r];
      }
    }
    __syncthreads();

    // row softmax (64 rows handled by the first 64 lanes), emit bf16 P
    if (tid < NPAD) {
      int m = tid;
      if (m < NTOK) {
        float mx = -1e30f;
        for (int n = 0; n < NTOK; ++n) mx = fmaxf(mx, sS[m * NPAD + n]);
        float sum = 0.0f;
        for (int n = 0; n < NTOK; ++n) sum += __expf(sS[m * NPAD + n] - mx);
        float inv = 1.0f / sum;
        for (int n = 0; n < NTOK; ++n)
          sP[m * NPAD + n] = (__bf16)(__expf(sS[m * NPAD + n] - mx) * inv);
        for (int n = NTOK; n < NPAD; ++n) sP[m * NPAD + n] = (__bf16)0.0f;
      } else {
        for (int n = 0; n < NPAD; ++n) sP[m * NPAD + n] = (__bf16)0.0f;
      }
    }
    __syncthreads();

    // O_h = P @ V_h : 4(m) x 2(n) tiles, one per wave, K=64 (2 WMMA steps)
    {
      int omt = wave >> 1, ont = wave & 1;
      v8f acc = {};
#pragma unroll
      for (int kt = 0; kt < 2; ++kt) {
        v16bf a  = load_a_frag(sP, NPAD, omt * 16, kt * 32, lane);
        v16bf bm = load_b_swz(sV, (h * 2 + ont) * 2 + kt, lane);
        acc = __builtin_amdgcn_wmma_f32_16x16x32_bf16(false, a, false, bm,
                                                      (short)0, acc, false, false);
      }
#pragma unroll
      for (int r = 0; r < 8; ++r) {
        int m = omt * 16 + r + (lane >> 4) * 8;
        int n = ont * 16 + (lane & 15);
        sO[m * CC + h * HD + n] = (__bf16)acc[r];
      }
    }
    __syncthreads();
  }

  // ---------------- Phase D: out = O @ proj_w + b, reverse-shift scatter -----
  for (int nb = 0; nb < CC / 64; ++nb) {
    for (int idx = tid; idx < CC * 64; idx += 256) {
      int k = idx >> 6, j = idx & 63;
      sW[b_swz_idx(k, j)] = (__bf16)proj_w[(size_t)k * CC + nb * 64 + j];
    }
    __syncthreads();

    v8f acc0 = {}, acc1 = {};
    v16bf a_c  = load_a_frag(sO, CC, mt * 16, 0, lane);
    v16bf b0_c = load_b_swz(sW, nt0, lane);
    v16bf b1_c = load_b_swz(sW, nt0 + 1, lane);
#pragma unroll
    for (int kt = 0; kt < 8; ++kt) {
      v16bf a_n, b0_n, b1_n;
      if (kt < 7) {
        a_n  = load_a_frag(sO, CC, mt * 16, (kt + 1) * 32, lane);
        b0_n = load_b_swz(sW, (kt + 1) * 4 + nt0, lane);
        b1_n = load_b_swz(sW, (kt + 1) * 4 + nt0 + 1, lane);
      }
      acc0 = __builtin_amdgcn_wmma_f32_16x16x32_bf16(false, a_c, false, b0_c,
                                                     (short)0, acc0, false, false);
      acc1 = __builtin_amdgcn_wmma_f32_16x16x32_bf16(false, a_c, false, b1_c,
                                                     (short)0, acc1, false, false);
      a_c = a_n; b0_c = b0_n; b1_c = b1_n;
    }

#pragma unroll
    for (int half = 0; half < 2; ++half) {
      const v8f& acc = half ? acc1 : acc0;
      int nt = nt0 + half;
#pragma unroll
      for (int r = 0; r < 8; ++r) {
        int m = mt * 16 + r + (lane >> 4) * 8;
        if (m < NTOK) {
          int gn = nb * 64 + nt * 16 + (lane & 15);
          int rr = m / WS, c = m - rr * WS;
          int hs = nh * WS + rr, ws = nw * WS + c;
          int hg = hs + SHIFT; if (hg >= HH)   hg -= HH;
          int wg = ws + SHIFT; if (wg >= WWID) wg -= WWID;
          out[(((size_t)bat * HH + hg) * WWID + wg) * CC + gn] = acc[r] + proj_b[gn];
        }
      }
    }
    __syncthreads();
  }
}

extern "C" void kernel_launch(void* const* d_in, const int* in_sizes, int n_in,
                              void* d_out, int out_size, void* d_ws, size_t ws_size,
                              hipStream_t stream) {
  (void)in_sizes; (void)n_in; (void)out_size; (void)d_ws; (void)ws_size;
  const float* x         = (const float*)d_in[0];
  const float* qkv_w     = (const float*)d_in[1];
  const float* qkv_b     = (const float*)d_in[2];
  const float* proj_w    = (const float*)d_in[3];
  const float* proj_b    = (const float*)d_in[4];
  const float* rel_bias  = (const float*)d_in[5];
  const float* attn_mask = (const float*)d_in[6];
  float* out = (float*)d_out;

  (void)hipFuncSetAttribute((const void*)swin_shifted_window_attn_kernel,
                            hipFuncAttributeMaxDynamicSharedMemorySize, SMEM_BYTES);

  dim3 grid(BB * NWIN);   // 1024 windows
  dim3 block(256);        // 8 wave32
  swin_shifted_window_attn_kernel<<<grid, block, SMEM_BYTES, stream>>>(
      x, qkv_w, qkv_b, proj_w, proj_b, rel_bias, attn_mask, out);
}
